// ScaledDotProductAttention_42588895707259
// MI455X (gfx1250) — matrix operand order repaired and checked by reading, not compiled, not run on hardware
//
#include <hip/hip_runtime.h>

#define CH   256
#define NTOK 4096
#define NB   4

typedef __attribute__((ext_vector_type(16))) __bf16       v16bf;
typedef __attribute__((ext_vector_type(8)))  float        v8f;
typedef __attribute__((ext_vector_type(4)))  float        v4f;
typedef __attribute__((ext_vector_type(8)))  unsigned int v8u;
typedef __attribute__((ext_vector_type(4)))  unsigned int v4u;

// ---- bf16 pack conversion -------------------------------------------------
static __device__ __forceinline__ unsigned int cvt2(float lo, float hi) {
#if __has_builtin(__builtin_amdgcn_cvt_pk_bf16_f32)
  return __builtin_bit_cast(unsigned int,
                            __builtin_amdgcn_cvt_pk_bf16_f32(lo, hi));
#else
  // round-half-up then v_perm_b32 picks the two high halfwords
  const unsigned int ulo = __float_as_uint(lo) + 0x8000u;
  const unsigned int uhi = __float_as_uint(hi) + 0x8000u;
  return __builtin_amdgcn_perm(uhi, ulo, 0x07060302u);
#endif
}
static __device__ __forceinline__ unsigned short cvt1(float f) {
  return (unsigned short)(cvt2(f, 0.0f) & 0xFFFFu);
}
static __device__ __forceinline__ v8f wmma_bf16(v8u a, v8u b, v8f c) {
  return __builtin_amdgcn_wmma_f32_16x16x32_bf16(
      false, __builtin_bit_cast(v16bf, a),
      false, __builtin_bit_cast(v16bf, b),
      (short)0, c, false, false);
}
static __device__ __forceinline__ v8u load_v8u(const void* p) {
  const v4u* q = (const v4u*)p;
  const v4u a = q[0];
  const v4u b = q[1];
  return __builtin_shufflevector(a, b, 0, 1, 2, 3, 4, 5, 6, 7);
}

// ---------------------------------------------------------------------------
// Kernel 0: one-shot weight conversion f32 -> packed bf16 (3 x 256x256).
// ---------------------------------------------------------------------------
__global__ __launch_bounds__(256, 1) void wcvt_kernel(
    const float* __restrict__ wq, const float* __restrict__ wk,
    const float* __restrict__ wv, unsigned int* __restrict__ Wb) {
  const int i   = blockIdx.x * 256 + threadIdx.x;  // dword index, 3*32768
  const int mat = i >> 15;
  const int off = i & 32767;
  const float* src = (mat == 0) ? wq : (mat == 1) ? wk : wv;
  Wb[i] = cvt2(src[2 * off], src[2 * off + 1]);
}

// ---------------------------------------------------------------------------
// Kernel 1: fused 1x1-conv (channel GEMM) + bias for Q/K/V (bf16 WMMA).
// Q,K -> token-major bf16 (B,N,C), Q pre-scaled by 1/sqrt(C).
// V   -> channel-major bf16 (B,C,N).
// ---------------------------------------------------------------------------
__global__ __launch_bounds__(128, 1) void qkv_proj_kernel(
    const float* __restrict__ xq, const float* __restrict__ xk,
    const float* __restrict__ xv,
    const unsigned short* __restrict__ Wb,
    const float* __restrict__ bq, const float* __restrict__ bk,
    const float* __restrict__ bv,
    unsigned short* __restrict__ Qb, unsigned short* __restrict__ Kb,
    unsigned short* __restrict__ Vt) {
  const int which = blockIdx.y;
  const float* x    = (which == 0) ? xq : (which == 1) ? xk : xv;
  const float* bias = (which == 0) ? bq : (which == 1) ? bk : bv;
  const unsigned short* w = Wb + (size_t)which * CH * CH;
  unsigned short* out = (which == 0) ? Qb : (which == 1) ? Kb : Vt;

  const int wave = threadIdx.x >> 5;
  const int lane = threadIdx.x & 31;
  const int b    = blockIdx.x >> 8;
  const int n0   = (blockIdx.x & 255) << 4;
  const int d0   = wave << 6;
  const int m    = lane & 15;
  const int half = lane >> 4;

  v8f acc[4];
#pragma unroll
  for (int s = 0; s < 4; ++s) acc[s] = (v8f){0, 0, 0, 0, 0, 0, 0, 0};

  for (int c0 = 0; c0 < CH; c0 += 32) {
    v8u A;
    const float* xb = x + ((size_t)b * CH + c0) * NTOK + n0 + m;
#pragma unroll
    for (int j = 0; j < 8; ++j) {
      const int kk = 2 * j + 16 * half;
      A[j] = cvt2(xb[(size_t)kk * NTOK], xb[(size_t)(kk + 1) * NTOK]);
    }
#pragma unroll
    for (int s = 0; s < 4; ++s) {
      const v8u Bf =
          load_v8u(w + (size_t)(d0 + 16 * s + m) * CH + c0 + 16 * half);
      acc[s] = wmma_bf16(A, Bf, acc[s]);
    }
  }

  const float qscale = 0.0625f;
#pragma unroll
  for (int s = 0; s < 4; ++s) {
    const int d = d0 + 16 * s + m;
    float bb = bias[d];
    float v[8];
#pragma unroll
    for (int i = 0; i < 8; ++i) {
      v[i] = acc[s][i] + bb;
      if (which == 0) v[i] *= qscale;
    }
    if (which == 2) {
      unsigned int* vo =
          (unsigned int*)(out + ((size_t)b * CH + d) * NTOK + n0 + 8 * half);
#pragma unroll
      for (int i = 0; i < 4; ++i) vo[i] = cvt2(v[2 * i], v[2 * i + 1]);
    } else {
#pragma unroll
      for (int i = 0; i < 8; ++i) {
        const int tok = n0 + i + 8 * half;
        out[((size_t)b * NTOK + tok) * CH + d] = cvt1(v[i]);
      }
    }
  }
}

// ---------------------------------------------------------------------------
// Kernel 2: cooperative flash attention.
// 256 threads = 8 waves; block covers 128 consecutive queries (16 per wave).
// Each 32-key step: K tile double-buffered in LDS, V tile single-buffered,
// all 8 waves share them (8x less K/V streaming than per-wave streaming).
// ---------------------------------------------------------------------------
#define KPITCH 264  // shorts per K row in LDS (528 B: 16B-aligned, 4m%64 banks)
#define VPITCH 40   // shorts per V row in LDS (80 B: 16B-aligned, 20m%64 banks)
#define PPITCH 36   // shorts per P row in LDS (72 B: 8B-aligned b64 gathers)

__global__ __launch_bounds__(256, 1) void flash_attn_kernel(
    const unsigned short* __restrict__ Qb, const unsigned short* __restrict__ Kb,
    const unsigned short* __restrict__ Vt, float* __restrict__ out) {
  __shared__ __align__(16) unsigned short ldsK[2][32 * KPITCH];  // 33.0 KB
  __shared__ __align__(16) unsigned short ldsV[256 * VPITCH];    // 20.5 KB
  __shared__ __align__(16) unsigned short ldsP[8][16 * PPITCH];  //  9.2 KB

  const int tid  = threadIdx.x;
  const int wave = tid >> 5;
  const int lane = tid & 31;
  const int b    = blockIdx.x >> 5;                      // 32 blocks per batch
  const int q0   = ((blockIdx.x & 31) << 7) + (wave << 4);
  const int m    = lane & 15;
  const int half = lane >> 4;

  const unsigned short* Kbase = Kb + (size_t)b * NTOK * CH;
  const unsigned short* Vbase = Vt + (size_t)b * CH * NTOK;

  // ---- preload Q A-fragments (8 K-chunks) into registers
  v8u Qf[8];
  {
    const unsigned short* qrow = Qb + ((size_t)b * NTOK + q0 + m) * CH;
#pragma unroll
    for (int c = 0; c < 8; ++c) {
      const v4u* p = (const v4u*)(qrow + c * 32 + 16 * half);
      Qf[c] = __builtin_shufflevector(p[0], p[1], 0, 1, 2, 3, 4, 5, 6, 7);
    }
  }

  // cooperative K tile stage: 32 rows x 512B -> pitch-528B LDS rows
  auto stageK = [&](int buf, int j0) {
#pragma unroll
    for (int c = 0; c < 4; ++c) {
      const int ch  = tid + c * 256;        // 16B chunk id, 1024 total
      const int row = ch >> 5;
      const int col = ch & 31;
      const v4u v = *(const v4u*)(Kbase + (size_t)(j0 + row) * CH + col * 8);
      *(v4u*)(&ldsK[buf][row * KPITCH + col * 8]) = v;
    }
  };

  float rmaxv[8], rsum[8];
#pragma unroll
  for (int i = 0; i < 8; ++i) { rmaxv[i] = -1e30f; rsum[i] = 0.0f; }
  v8f O[16];
#pragma unroll
  for (int t = 0; t < 16; ++t) O[t] = (v8f){0, 0, 0, 0, 0, 0, 0, 0};

  stageK(0, 0);
  __syncthreads();

  int buf = 0;
  for (int j0 = 0; j0 < NTOK; j0 += 32, buf ^= 1) {
    // ---- issue this step's V tile loads early (each thread: one channel row)
    v4u vst[4];
    {
      const v4u* vsrc = (const v4u*)(Vbase + (size_t)tid * NTOK + j0);
#pragma unroll
      for (int i = 0; i < 4; ++i) vst[i] = vsrc[i];
    }

    // ---- S = Q.K^T from shared K tile (two 16x16 f32 tiles)
    v8f S0 = (v8f){0, 0, 0, 0, 0, 0, 0, 0};
    v8f S1 = (v8f){0, 0, 0, 0, 0, 0, 0, 0};
#pragma unroll
    for (int c0 = 0; c0 < CH; c0 += 32) {
      const unsigned short* kr0 = &ldsK[buf][m * KPITCH + c0 + 16 * half];
      const v8u B0 = load_v8u(kr0);
      const v8u B1 = load_v8u(kr0 + 16 * KPITCH);
      S0 = wmma_bf16(Qf[c0 >> 5], B0, S0);
      S1 = wmma_bf16(Qf[c0 >> 5], B1, S1);
    }

    // ---- stage next K tile into the other buffer
    if (j0 + 32 < NTOK) stageK(buf ^ 1, j0 + 32);

    // ---- commit V tile to LDS
#pragma unroll
    for (int i = 0; i < 4; ++i)
      *(v4u*)(&ldsV[tid * VPITCH + i * 8]) = vst[i];

    // ---- online softmax (rows live in 16-lane halves of the C/D layout)
    float pr[8];
#pragma unroll
    for (int i = 0; i < 8; ++i) pr[i] = fmaxf(S0[i], S1[i]);
#pragma unroll
    for (int mk = 1; mk < 16; mk <<= 1)
#pragma unroll
      for (int i = 0; i < 8; ++i)
        pr[i] = fmaxf(pr[i], __shfl_xor(pr[i], mk, 32));

    float scale[8];
    bool resc = false;
#pragma unroll
    for (int i = 0; i < 8; ++i) {
      const float mn = fmaxf(rmaxv[i], pr[i]);
      scale[i] = __expf(rmaxv[i] - mn);
      rmaxv[i] = mn;
      resc |= (scale[i] < 1.0f);
    }
    float ps[8];
#pragma unroll
    for (int i = 0; i < 8; ++i) {
      S0[i] = __expf(S0[i] - rmaxv[i]);
      S1[i] = __expf(S1[i] - rmaxv[i]);
      ps[i] = S0[i] + S1[i];
    }
#pragma unroll
    for (int mk = 1; mk < 16; mk <<= 1)
#pragma unroll
      for (int i = 0; i < 8; ++i) ps[i] += __shfl_xor(ps[i], mk, 32);
#pragma unroll
    for (int i = 0; i < 8; ++i) rsum[i] = rsum[i] * scale[i] + ps[i];

    if (__ballot(resc)) {  // skip 128 muls when no row max moved
#pragma unroll
      for (int t = 0; t < 16; ++t)
#pragma unroll
        for (int i = 0; i < 8; ++i) O[t][i] *= scale[i];
    }

    // ---- P (C-layout) -> A-fragment layout via per-wave LDS scratch
    {
      unsigned short* pp = ldsP[wave];
#pragma unroll
      for (int i = 0; i < 8; ++i) {
        const int mr = i + 8 * half;
        pp[mr * PPITCH + m]      = cvt1(S0[i]);
        pp[mr * PPITCH + 16 + m] = cvt1(S1[i]);
      }
    }

    __syncthreads();  // V + K[buf^1] staged; everyone done reading K[buf]

    v8u P;
    {
      const unsigned int* pb =
          (const unsigned int*)ldsP[wave] + m * (PPITCH / 2) + 4 * half;
      typedef __attribute__((ext_vector_type(2))) unsigned int v2u;
      const v2u p0 = *(const v2u*)(pb);
      const v2u p1 = *(const v2u*)(pb + 2);
      const v2u p2 = *(const v2u*)(pb + 8);
      const v2u p3 = *(const v2u*)(pb + 10);
      const v4u lo = __builtin_shufflevector(p0, p1, 0, 1, 2, 3);
      const v4u hi = __builtin_shufflevector(p2, p3, 0, 1, 2, 3);
      P = __builtin_shufflevector(lo, hi, 0, 1, 2, 3, 4, 5, 6, 7);
    }

    // ---- O += P.V from shared V tile
#pragma unroll
    for (int t = 0; t < 16; ++t) {
      const unsigned short* vr = &ldsV[(t * 16 + m) * VPITCH + 8 * half];
      const v8u Vf = __builtin_shufflevector(*(const v4u*)vr, *(const v4u*)(vr + 16), 0, 1, 2, 3, 4, 5, 6, 7);
      O[t] = wmma_bf16(P, Vf, O[t]);
    }

    __syncthreads();  // all AV reads of ldsV done before next overwrite
  }

  // ---- epilogue: normalize, store (B,C,H,W) fp32 with 16B vector stores
  float inv[8];
#pragma unroll
  for (int i = 0; i < 8; ++i) inv[i] = 1.0f / rsum[i];
  float* ob = out + (size_t)b * CH * NTOK;
#pragma unroll
  for (int t = 0; t < 16; ++t) {
    const int d = t * 16 + m;
    float* p = ob + (size_t)d * NTOK + q0 + 8 * half;
    v4f lo, hi;
#pragma unroll
    for (int i = 0; i < 4; ++i) {
      lo[i] = O[t][i] * inv[i];
      hi[i] = O[t][i + 4] * inv[i + 4];
    }
    *(v4f*)(p)     = lo;
    *(v4f*)(p + 4) = hi;
  }
}

// ---------------------------------------------------------------------------
extern "C" void kernel_launch(void* const* d_in, const int* in_sizes, int n_in,
                              void* d_out, int out_size, void* d_ws,
                              size_t ws_size, hipStream_t stream) {
  const float* xq = (const float*)d_in[0];
  const float* xk = (const float*)d_in[1];
  const float* xv = (const float*)d_in[2];
  const float* wq = (const float*)d_in[3];
  const float* bq = (const float*)d_in[4];
  const float* wk = (const float*)d_in[5];
  const float* bk = (const float*)d_in[6];
  const float* wv = (const float*)d_in[7];
  const float* bv = (const float*)d_in[8];

  const size_t tsz = (size_t)NB * NTOK * CH;
  unsigned short* Qb = (unsigned short*)d_ws;
  unsigned short* Kb = Qb + tsz;
  unsigned short* Vt = Kb + tsz;
  unsigned short* Wb = Vt + tsz;

  wcvt_kernel<<<dim3(384), 256, 0, stream>>>(wq, wk, wv, (unsigned int*)Wb);

  qkv_proj_kernel<<<dim3(NB * (NTOK / 16), 3, 1), 128, 0, stream>>>(
      xq, xk, xv, Wb, bq, bk, bv, Qb, Kb, Vt);

  // 128 blocks x 256 threads: each block = 128 queries sharing K/V tiles
  flash_attn_kernel<<<dim3(NB * (NTOK / 128), 1, 1), 256, 0, stream>>>(
      Qb, Kb, Vt, (float*)d_out);
}
